// InputEmbedding_47278999995065
// MI455X (gfx1250) — matrix operand order repaired and checked by reading, not compiled, not run on hardware
//
#include <hip/hip_runtime.h>
#include <hip/hip_bf16.h>
#include <cstdint>
#include <cstddef>

typedef __bf16 bf16;
typedef __attribute__((ext_vector_type(16))) __bf16 v16bf;
typedef __attribute__((ext_vector_type(8)))  __bf16 v8bf;
typedef __attribute__((ext_vector_type(8)))  float  v8f;
typedef __attribute__((ext_vector_type(4)))  int    v4i;

typedef __attribute__((address_space(1))) v4i gv4i;   // global int4
typedef __attribute__((address_space(3))) v4i lv4i;   // LDS int4

#define BN_EPS 1e-5f
#define KSTEP  64
#define LDSPAD 8   // bf16 elements of row padding to spread LDS banks

#if defined(__has_builtin)
#if __has_builtin(__builtin_amdgcn_global_load_async_to_lds_b128) && \
    __has_builtin(__builtin_amdgcn_s_wait_asynccnt)
#define USE_ASYNC_LDS 1
#endif
#endif

// 16-byte async copy global -> LDS (ASYNCcnt tracked); no VGPR round trip.
__device__ __forceinline__ void async_cp16(const bf16* g, bf16* l) {
#ifdef USE_ASYNC_LDS
  __builtin_amdgcn_global_load_async_to_lds_b128(
      (gv4i*)g, (lv4i*)l, 0, 0);
#else
  (void)g; (void)l;
#endif
}

// =====================================================================
// GEMM:  C(M,N) bf16  =  A(M,K) bf16  @  W(N,K)^T   (fp32 WMMA accum)
// Block = 256 threads = 8 waves. Block tile 256x64; wave tile 32x64.
// B tile (64 cols x 64 k) staged through double-buffered LDS via
// GLOBAL_LOAD_ASYNC_TO_LDS (fallback: regular loads + ds_store).
// Requires: M % 256 == 0, N % 64 == 0, K % 64 == 0.
// =====================================================================
__global__ __launch_bounds__(256) void gemm_bf16_wmma(
    const bf16* __restrict__ A, const bf16* __restrict__ W,
    bf16* __restrict__ C, int M, int N, int K)
{
  __shared__ __align__(64) bf16 Bs[2][64][KSTEP + LDSPAD];

  const int tid  = threadIdx.x;
  const int lane = tid & 31;
  const int wave = tid >> 5;
  const int m0 = blockIdx.x * 256 + wave * 32;
  const int n0 = blockIdx.y * 64;

  // ---- B tile staging: 64 cols x 64 k = 8KB; each thread moves 32B ----
  const int scol = tid >> 2;          // 0..63
  const int skof = (tid & 3) << 4;    // 0,16,32,48 (bf16 elements)
  const bf16* Wst = W + (size_t)(n0 + scol) * K + skof;

  // ---- A fragment addressing (16x32 bf16 wave32 layout) ----
  //   lanes 0-15: M=lane,  K groups {0..7, 16..23}
  //   lanes 16-31: M=lane-16, K groups {8..15, 24..31}
  const int arow = m0 + (lane & 15);
  const int akb  = (lane >> 4) << 3;               // 0 or 8
  const bf16* Ap0 = A + (size_t)arow * K;          // rows m0+0..15
  const bf16* Ap1 = Ap0 + (size_t)16 * K;          // rows m0+16..31

  // ---- B fragment addressing (32x16): col=lane&15, K groups {0..7,16..23}/{8..15,24..31} ----
  const int bcol = lane & 15;
  const int bkb  = (lane >> 4) << 3;               // 0 or 8

  v8f acc[2][4] = {};

  // ---- prologue: stage k-tile 0 into buffer 0 ----
#ifdef USE_ASYNC_LDS
  async_cp16(Wst,     &Bs[0][scol][skof]);
  async_cp16(Wst + 8, &Bs[0][scol][skof + 8]);
  __builtin_amdgcn_s_wait_asynccnt(0);
#else
  {
    v8bf s0 = *(const v8bf*)(Wst);
    v8bf s1 = *(const v8bf*)(Wst + 8);
    *(v8bf*)(&Bs[0][scol][skof])     = s0;
    *(v8bf*)(&Bs[0][scol][skof + 8]) = s1;
  }
#endif
  __syncthreads();

  const int TILES = K / KSTEP;
  for (int t = 0; t < TILES; ++t) {
    const int k0  = t * KSTEP;
    const int buf = t & 1;

#ifdef USE_ASYNC_LDS
    // Issue async writes to the other buffer immediately: every wave's reads
    // of buf^1 (from iteration t-1) completed before the barrier that ended
    // iteration t-1, so this is race-free with a single barrier per tile.
    if (t + 1 < TILES) {
      const bf16* Wn = Wst + k0 + KSTEP;
      async_cp16(Wn,     &Bs[buf ^ 1][scol][skof]);
      async_cp16(Wn + 8, &Bs[buf ^ 1][scol][skof + 8]);
    }
#else
    v8bf nt0 = {}, nt1 = {};
    if (t + 1 < TILES) {
      const bf16* Wn = Wst + k0 + KSTEP;
      nt0 = *(const v8bf*)(Wn);
      nt1 = *(const v8bf*)(Wn + 8);
    }
#endif

    #pragma unroll
    for (int kc = 0; kc < 2; ++kc) {
      const int kk = k0 + kc * 32;
      v8bf a0lo = *(const v8bf*)(Ap0 + kk + akb);
      v8bf a0hi = *(const v8bf*)(Ap0 + kk + 16 + akb);
      v8bf a1lo = *(const v8bf*)(Ap1 + kk + akb);
      v8bf a1hi = *(const v8bf*)(Ap1 + kk + 16 + akb);
      v16bf a0 = __builtin_shufflevector(a0lo, a0hi,
          0,1,2,3,4,5,6,7,8,9,10,11,12,13,14,15);
      v16bf a1 = __builtin_shufflevector(a1lo, a1hi,
          0,1,2,3,4,5,6,7,8,9,10,11,12,13,14,15);

      #pragma unroll
      for (int j = 0; j < 4; ++j) {
        const bf16* bp = &Bs[buf][j * 16 + bcol][kc * 32 + bkb];
        v8bf blo = *(const v8bf*)(bp);
        v8bf bhi = *(const v8bf*)(bp + 16);
        v16bf b = __builtin_shufflevector(blo, bhi,
            0,1,2,3,4,5,6,7,8,9,10,11,12,13,14,15);
        acc[0][j] = __builtin_amdgcn_wmma_f32_16x16x32_bf16(
            false, a0, false, b, (short)0, acc[0][j], false, false);
        acc[1][j] = __builtin_amdgcn_wmma_f32_16x16x32_bf16(
            false, a1, false, b, (short)0, acc[1][j], false, false);
      }
    }

#ifdef USE_ASYNC_LDS
    if (t + 1 < TILES) {
      __builtin_amdgcn_s_wait_asynccnt(0);  // my async writes landed in LDS
      __syncthreads();                       // everyone's landed; reads done
    }
#else
    __syncthreads();                 // all waves done reading Bs[buf]
    if (t + 1 < TILES) {
      *(v8bf*)(&Bs[buf ^ 1][scol][skof])     = nt0;
      *(v8bf*)(&Bs[buf ^ 1][scol][skof + 8]) = nt1;
      __syncthreads();               // next tile visible to all waves
    }
#endif
  }

  // C/D layout: VGPR r -> row = m0 + r (lanes 0-15) or m0 + 8 + r (lanes 16-31)
  const int crow = m0 + ((lane >> 4) << 3);
  const int ccol = n0 + (lane & 15);
  #pragma unroll
  for (int mi = 0; mi < 2; ++mi) {
    #pragma unroll
    for (int r = 0; r < 8; ++r) {
      size_t o = (size_t)(crow + mi * 16 + r) * N + ccol;
      C[o     ] = (bf16)acc[mi][0][r];
      C[o + 16] = (bf16)acc[mi][1][r];
      C[o + 32] = (bf16)acc[mi][2][r];
      C[o + 48] = (bf16)acc[mi][3][r];
    }
  }
}

// ========================= small helpers ============================

__global__ void cvt_f32_bf16(const float* __restrict__ in, bf16* __restrict__ out, int n) {
  int i = blockIdx.x * blockDim.x + threadIdx.x;
  if (i < n) out[i] = (bf16)in[i];
}

__global__ void zero_f32(float* __restrict__ p, int n) {
  int i = blockIdx.x * blockDim.x + threadIdx.x;
  if (i < n) p[i] = 0.0f;
}

// First tiny matmul: (BN,3) @ (O,3)^T -> bf16 (BN,O)
__global__ void gemm_xyz(const float* __restrict__ xyz, const float* __restrict__ W1,
                         bf16* __restrict__ out, int Mtot, int O) {
  int t = blockIdx.x * blockDim.x + threadIdx.x;
  if (t >= Mtot * O) return;
  int o = t % O;
  int r = t / O;
  float s = xyz[r*3+0]*W1[o*3+0] + xyz[r*3+1]*W1[o*3+1] + xyz[r*3+2]*W1[o*3+2];
  out[t] = (bf16)s;
}

// per-channel partial sums (sum, sumsq) with atomics; sums has 2*C floats
__global__ void stats_partial(const bf16* __restrict__ Z, float* __restrict__ sums,
                              int M, int C) {
  int c = blockIdx.x * blockDim.x + threadIdx.x;
  if (c >= C) return;
  int chunks = gridDim.y;
  int rows_per = (M + chunks - 1) / chunks;
  int r0 = blockIdx.y * rows_per;
  int r1 = min(M, r0 + rows_per);
  float s = 0.0f, ss = 0.0f;
  for (int r = r0; r < r1; ++r) {
    float v = (float)Z[(size_t)r * C + c];
    s += v; ss += v * v;
  }
  atomicAdd(&sums[c], s);
  atomicAdd(&sums[C + c], ss);
}

// fold BN into y = x*scale + shift
__global__ void finalize_stats(const float* __restrict__ sums,
                               const float* __restrict__ g, const float* __restrict__ bta,
                               float* __restrict__ scale, float* __restrict__ shift,
                               int C, float invM) {
  int c = blockIdx.x * blockDim.x + threadIdx.x;
  if (c >= C) return;
  float mean = sums[c] * invM;
  float var  = sums[C + c] * invM - mean * mean;
  float rs   = rsqrtf(var + BN_EPS);
  float sc   = rs * g[c];
  scale[c] = sc;
  shift[c] = bta[c] - mean * sc;
}

__global__ void bn_relu_apply(const bf16* __restrict__ Z,
                              const float* __restrict__ scale, const float* __restrict__ shift,
                              bf16* __restrict__ out, size_t total, int C) {
  size_t t = (size_t)blockIdx.x * blockDim.x + threadIdx.x;
  if (t >= total) return;
  int c = (int)(t % C);
  float v = (float)Z[t] * scale[c] + shift[c];
  out[t] = (bf16)fmaxf(v, 0.0f);
}

// apply BN+ReLU then max over k (rows grouped by Kn) -> bf16 (Mout,C)
__global__ void bn_relu_max(const bf16* __restrict__ Z,
                            const float* __restrict__ scale, const float* __restrict__ shift,
                            bf16* __restrict__ out, int Mout, int Kn, int C) {
  size_t t = (size_t)blockIdx.x * blockDim.x + threadIdx.x;
  if (t >= (size_t)Mout * C) return;
  int c = (int)(t % C);
  size_t r = t / C;
  float m = 0.0f;
  for (int k = 0; k < Kn; ++k) {
    float v = (float)Z[((r * Kn) + k) * (size_t)C + c] * scale[c] + shift[c];
    m = fmaxf(m, v);           // max of relu == relu of max (relu monotonic, >=0)
  }
  out[t] = (bf16)m;
}

// final: apply BN+ReLU, max over k, write f32 transposed (b, c, n)
__global__ void bn_relu_max_tr(const bf16* __restrict__ Z,
                               const float* __restrict__ scale, const float* __restrict__ shift,
                               float* __restrict__ out, int B, int N, int Kn, int C) {
  size_t t = (size_t)blockIdx.x * blockDim.x + threadIdx.x;
  if (t >= (size_t)B * N * C) return;
  int c = (int)(t % C);
  size_t r = t / C;            // r = b*N + n
  int n = (int)(r % N);
  int b = (int)(r / N);
  float m = 0.0f;
  for (int k = 0; k < Kn; ++k) {
    float v = (float)Z[((r * Kn) + k) * (size_t)C + c] * scale[c] + shift[c];
    m = fmaxf(m, v);
  }
  out[((size_t)b * C + c) * N + n] = m;
}

// =========================== kNN (k=8) ==============================
// one block of 256 threads handles 256 points of one batch; batch xyz in LDS
__global__ __launch_bounds__(256) void knn_kernel(const float* __restrict__ xyz,
                                                  int* __restrict__ idx, int N) {
  __shared__ float sx[1024], sy[1024], sz[1024];
  int b = blockIdx.y;
  const float* base = xyz + (size_t)b * N * 3;
  for (int i = threadIdx.x; i < N; i += blockDim.x) {
    sx[i] = base[i*3+0]; sy[i] = base[i*3+1]; sz[i] = base[i*3+2];
  }
  __syncthreads();
  int n = blockIdx.x * blockDim.x + threadIdx.x;
  if (n >= N) return;
  float px = sx[n], py = sy[n], pz = sz[n];
  float bestd[8]; int besti[8];
  #pragma unroll
  for (int t = 0; t < 8; ++t) { bestd[t] = INFINITY; besti[t] = 0; }
  for (int j = 0; j < N; ++j) {
    float dx = sx[j]-px, dy = sy[j]-py, dz = sz[j]-pz;
    float d = dx*dx + dy*dy + dz*dz;
    if (d < bestd[7]) {
      int p = 7;
      while (p > 0 && bestd[p-1] > d) { bestd[p]=bestd[p-1]; besti[p]=besti[p-1]; --p; }
      bestd[p] = d; besti[p] = j;
    }
  }
  int* o = idx + ((size_t)b * N + n) * 8;
  #pragma unroll
  for (int t = 0; t < 8; ++t) o[t] = besti[t];
}

// group: X[(b,n,k), 0..2D) = concat(feat[b,idx]-feat[b,n], feat[b,n])
__global__ void group_kernel(const bf16* __restrict__ feat, const int* __restrict__ idx,
                             bf16* __restrict__ out, int N, int Kn, int D, size_t total) {
  size_t t = (size_t)blockIdx.x * blockDim.x + threadIdx.x;
  if (t >= total) return;
  int C2 = 2 * D;
  size_t r = t / C2;
  int c = (int)(t % C2);
  size_t bn = r / Kn;               // b*N + n
  size_t bbase = (bn / N) * (size_t)N;
  float v;
  if (c < D) {
    int j = idx[r];
    v = (float)feat[(bbase + (size_t)j) * D + c] - (float)feat[bn * D + c];
  } else {
    v = (float)feat[bn * D + (c - D)];
  }
  out[t] = (bf16)v;
}

// ============================ driver ================================

static void bn_stage_stats(const bf16* Z, int M, int C,
                           const float* g, const float* b,
                           float* sums, float* scale, float* shift,
                           hipStream_t stream) {
  zero_f32<<<(2*C + 255)/256, 256, 0, stream>>>(sums, 2*C);
  dim3 gs((C + 255)/256, 64);
  stats_partial<<<gs, 256, 0, stream>>>(Z, sums, M, C);
  finalize_stats<<<(C + 255)/256, 256, 0, stream>>>(sums, g, b, scale, shift, C, 1.0f/(float)M);
}

extern "C" void kernel_launch(void* const* d_in, const int* in_sizes, int n_in,
                              void* d_out, int out_size, void* d_ws, size_t ws_size,
                              hipStream_t stream)
{
  (void)in_sizes; (void)n_in; (void)out_size; (void)ws_size;
  const float* xyz = (const float*)d_in[0];
  const float* W1  = (const float*)d_in[1];
  const float* g1  = (const float*)d_in[2];
  const float* b1  = (const float*)d_in[3];
  const float* W2  = (const float*)d_in[4];
  const float* g2  = (const float*)d_in[5];
  const float* b2  = (const float*)d_in[6];
  const float* WA1 = (const float*)d_in[7];
  const float* gA1 = (const float*)d_in[8];
  const float* bA1 = (const float*)d_in[9];
  const float* WA2 = (const float*)d_in[10];
  const float* gA2 = (const float*)d_in[11];
  const float* bA2 = (const float*)d_in[12];
  const float* WB1 = (const float*)d_in[13];
  const float* gB1 = (const float*)d_in[14];
  const float* bB1 = (const float*)d_in[15];
  const float* WB2 = (const float*)d_in[16];
  const float* gB2 = (const float*)d_in[17];
  const float* bB2 = (const float*)d_in[18];
  float* out = (float*)d_out;

  const int B = 8, N = 1024, KN = 8;
  const int D1 = 64, D2 = 256, D3 = 512, D4 = 1024;
  const int BN  = B * N;      // 8192
  const int BNK = BN * KN;    // 65536

  char* ws = (char*)d_ws;
  size_t off = 0;
  auto walloc = [&](size_t bytes) -> char* {
    char* p = ws + off; off += (bytes + 255) & ~(size_t)255; return p;
  };
  bf16* Zb   = (bf16*)walloc((size_t)BNK * D4 * 2);   // 128 MB GEMM output ping
  bf16* Xb   = (bf16*)walloc((size_t)BNK * D4 * 2);   // 128 MB GEMM input pong
  bf16* w2b  = (bf16*)walloc((size_t)D2 * D1 * 2);
  bf16* wa1b = (bf16*)walloc((size_t)D3 * D3 * 2);
  bf16* wa2b = (bf16*)walloc((size_t)D3 * D3 * 2);
  bf16* wb1b = (bf16*)walloc((size_t)D4 * D4 * 2);
  bf16* wb2b = (bf16*)walloc((size_t)D4 * D4 * 2);
  bf16* h1   = (bf16*)walloc((size_t)BN * D1 * 2);
  bf16* h1b  = (bf16*)walloc((size_t)BN * D1 * 2);
  bf16* z2   = (bf16*)walloc((size_t)BN * D2 * 2);
  bf16* h2b  = (bf16*)walloc((size_t)BN * D2 * 2);
  bf16* l0   = (bf16*)walloc((size_t)BN * D3 * 2);
  int*  idx  = (int*) walloc((size_t)BNK * 4);
  float* sums  = (float*)walloc(2 * D4 * 4);
  float* scale = (float*)walloc(D4 * 4);
  float* shift = (float*)walloc(D4 * 4);

  // ---- weight conversions to bf16 ----
  cvt_f32_bf16<<<(D2*D1 + 255)/256, 256, 0, stream>>>(W2, w2b, D2*D1);
  cvt_f32_bf16<<<(D3*D3 + 255)/256, 256, 0, stream>>>(WA1, wa1b, D3*D3);
  cvt_f32_bf16<<<(D3*D3 + 255)/256, 256, 0, stream>>>(WA2, wa2b, D3*D3);
  cvt_f32_bf16<<<(D4*D4 + 255)/256, 256, 0, stream>>>(WB1, wb1b, D4*D4);
  cvt_f32_bf16<<<(D4*D4 + 255)/256, 256, 0, stream>>>(WB2, wb2b, D4*D4);

  // ---- stage 1: h = bn_relu(xyz @ W1^T)  (BN x 64) ----
  gemm_xyz<<<(BN*D1 + 255)/256, 256, 0, stream>>>(xyz, W1, h1, BN, D1);
  bn_stage_stats(h1, BN, D1, g1, b1, sums, scale, shift, stream);
  bn_relu_apply<<<(int)(((size_t)BN*D1 + 255)/256), 256, 0, stream>>>(h1, scale, shift, h1b, (size_t)BN*D1, D1);

  // ---- stage 2: h = bn_relu(h @ W2^T)  (BN x 256) ----
  gemm_bf16_wmma<<<dim3(BN/256, D2/64), 256, 0, stream>>>(h1b, w2b, z2, BN, D2, D1);
  bn_stage_stats(z2, BN, D2, g2, b2, sums, scale, shift, stream);
  bn_relu_apply<<<(int)(((size_t)BN*D2 + 255)/256), 256, 0, stream>>>(z2, scale, shift, h2b, (size_t)BN*D2, D2);

  // ---- kNN (shared by both group stages; depends only on xyz) ----
  knn_kernel<<<dim3(N/256, B), 256, 0, stream>>>(xyz, idx, N);

  // ---- group 0: X0 (BNK x 512) ----
  {
    size_t total = (size_t)BNK * (2*D2);
    group_kernel<<<(int)((total + 255)/256), 256, 0, stream>>>(h2b, idx, Xb, N, KN, D2, total);
  }

  // ---- local_op A ----
  gemm_bf16_wmma<<<dim3(BNK/256, D3/64), 256, 0, stream>>>(Xb, wa1b, Zb, BNK, D3, D3);
  bn_stage_stats(Zb, BNK, D3, gA1, bA1, sums, scale, shift, stream);
  bn_relu_apply<<<(int)(((size_t)BNK*D3 + 255)/256), 256, 0, stream>>>(Zb, scale, shift, Xb, (size_t)BNK*D3, D3);

  gemm_bf16_wmma<<<dim3(BNK/256, D3/64), 256, 0, stream>>>(Xb, wa2b, Zb, BNK, D3, D3);
  bn_stage_stats(Zb, BNK, D3, gA2, bA2, sums, scale, shift, stream);
  bn_relu_max<<<(int)(((size_t)BN*D3 + 255)/256), 256, 0, stream>>>(Zb, scale, shift, l0, BN, KN, D3);

  // ---- group 1: X1 (BNK x 1024) ----
  {
    size_t total = (size_t)BNK * (2*D3);
    group_kernel<<<(int)((total + 255)/256), 256, 0, stream>>>(l0, idx, Xb, N, KN, D3, total);
  }

  // ---- local_op B ----
  gemm_bf16_wmma<<<dim3(BNK/256, D4/64), 256, 0, stream>>>(Xb, wb1b, Zb, BNK, D4, D4);
  bn_stage_stats(Zb, BNK, D4, gB1, bB1, sums, scale, shift, stream);
  bn_relu_apply<<<(int)(((size_t)BNK*D4 + 255)/256), 256, 0, stream>>>(Zb, scale, shift, Xb, (size_t)BNK*D4, D4);

  gemm_bf16_wmma<<<dim3(BNK/256, D4/64), 256, 0, stream>>>(Xb, wb2b, Zb, BNK, D4, D4);
  bn_stage_stats(Zb, BNK, D4, gB2, bB2, sums, scale, shift, stream);
  bn_relu_max_tr<<<(int)(((size_t)BN*D4 + 255)/256), 256, 0, stream>>>(Zb, scale, shift, out, B, N, KN, D4);
}
